// MaskedCrossAtten_72765335929383
// MI455X (gfx1250) — matrix operand order repaired and checked by reading, not compiled, not run on hardware
//
#include <hip/hip_runtime.h>
#include <cstdint>

// ---------------- problem constants ----------------
#define BS   4
#define QD   1024
#define T1   2048
#define T2   2048
#define NH   16
#define HD   64          // head dim
#define SCALE 0.125f     // 1/sqrt(64)

#define PADW 8
#define PADX 8
#define PADK 8
#define PADV 8
#define PADP 8

typedef __attribute__((ext_vector_type(16))) __bf16 v16bf;
typedef __attribute__((ext_vector_type(4)))  __bf16 v4bf;
typedef __attribute__((ext_vector_type(2)))  __bf16 v2bf;
typedef __attribute__((ext_vector_type(8)))  float  v8f;

union Frag { uint4 u[2]; v16bf v; };

// packed f32x2 -> bf16x2: hardware builtin if present, else native fptrunc
// (backend lowers to v_cvt_pk_bf16_f32-class ops on gfx1250)
__device__ __forceinline__ v2bf cvt2(float a, float b) {
#if __has_builtin(__builtin_amdgcn_cvt_pk_bf16_f32)
    return __builtin_amdgcn_cvt_pk_bf16_f32(a, b);
#else
    v2bf r; r[0] = (__bf16)a; r[1] = (__bf16)b; return r;
#endif
}

__device__ __forceinline__ uint4 pack8_bf16(const float* s) {
    union { uint4 u; v2bf h[4]; } p;
    #pragma unroll
    for (int i = 0; i < 4; ++i) p.h[i] = cvt2(s[2 * i], s[2 * i + 1]);
    return p.u;
}

__device__ __forceinline__ v8f bf16mma(v16bf a, v16bf b, v8f c) {
    return __builtin_amdgcn_wmma_f32_16x16x32_bf16(
        false, a, false, b, (short)0, c, false, false);
}

// CDNA5 async DMA: global -> LDS, 16 bytes per lane, tracked by ASYNCcnt.
// VDST = LDS byte address (low 32 bits of flat shared pointer), VADDR = global VA.
__device__ __forceinline__ void async_b128(void* lds, const void* g) {
    unsigned loff = (unsigned)(size_t)lds;
    unsigned long long ga = (unsigned long long)(size_t)g;
    asm volatile("global_load_async_to_lds_b128 %0, %1, off"
                 :: "v"(loff), "v"(ga) : "memory");
}
__device__ __forceinline__ void wait_async0() {
    asm volatile("s_wait_asynccnt 0" ::: "memory");
}

// =====================================================================
// Projection:  out(o,t) = W(o,:) . X(:,t) + bias(o)   per batch
// Block 256 thr (8 waves). Tile 64(o) x 128(t), K-step 32.
//   MODE 0: Q -> (b,h,t,d) * SCALE
//   MODE 1: K -> (b,h,t,d)
//   MODE 2: V -> (b,h,d,t)
// =====================================================================
template <int MODE>
__global__ __launch_bounds__(256)
void proj_wmma_kernel(const float* __restrict__ W, const float* __restrict__ X,
                      const float* __restrict__ bias, __bf16* __restrict__ out)
{
    const int obase = blockIdx.x * 64;     // h == blockIdx.x (64 = HD)
    const int tbase = blockIdx.y * 128;
    const int b     = blockIdx.z;
    const int tid   = threadIdx.x;
    const int lane  = tid & 31;
    const int wave  = tid >> 5;
    const int wm    = wave & 3;   // o sub-strip (16 rows)
    const int wn    = wave >> 2;  // t half (64 cols)

    __shared__ __align__(16) __bf16 Wl[64][32 + PADW];    // (o , c)
    __shared__ __align__(16) __bf16 Xl[128][32 + PADX];   // (t , c)  transposed

    v8f acc[4] = {};
    const int n    = lane & 15;
    const int rowb = (lane < 16) ? 0 : 16;   // B fragment K-dim base
    const int kbA  = (lane < 16) ? 0 : 8;    // A fragment K base
    const int Moff = (lane < 16) ? 0 : 8;    // C fragment M offset

    for (int c0 = 0; c0 < QD; c0 += 32) {
        __syncthreads();
        {   // W tile 64x32: 8 contiguous c per thread -> one b128 LDS store
            int r = tid >> 2, cg = (tid & 3) * 8;
            const float* src = &W[(size_t)(obase + r) * QD + c0 + cg];
            *(uint4*)&Wl[r][cg] = pack8_bf16(src);
        }
        {   // X tile 32c x 128t, transposed into Xl[t][c] via 4x4 micro-block
            int c = (tid >> 5) * 4;   // 0..28
            int t = lane * 4;         // 0..124
            const float* base = &X[((size_t)b * QD + c0 + c) * (size_t)T1 + tbase + t];
            float4 r0 = *(const float4*)(base);
            float4 r1 = *(const float4*)(base + T1);
            float4 r2 = *(const float4*)(base + 2 * T1);
            float4 r3 = *(const float4*)(base + 3 * T1);
            union { v2bf h[2]; v4bf v; } q0, q1, q2, q3;
            q0.h[0] = cvt2(r0.x, r1.x); q0.h[1] = cvt2(r2.x, r3.x);
            q1.h[0] = cvt2(r0.y, r1.y); q1.h[1] = cvt2(r2.y, r3.y);
            q2.h[0] = cvt2(r0.z, r1.z); q2.h[1] = cvt2(r2.z, r3.z);
            q3.h[0] = cvt2(r0.w, r1.w); q3.h[1] = cvt2(r2.w, r3.w);
            *(v4bf*)&Xl[t + 0][c] = q0.v;
            *(v4bf*)&Xl[t + 1][c] = q1.v;
            *(v4bf*)&Xl[t + 2][c] = q2.v;
            *(v4bf*)&Xl[t + 3][c] = q3.v;
        }
        __syncthreads();

        Frag a;
        {
            int row = wm * 16 + n;
            a.u[0] = *(const uint4*)&Wl[row][kbA];
            a.u[1] = *(const uint4*)&Wl[row][kbA + 16];
        }
        #pragma unroll
        for (int j = 0; j < 4; ++j) {
            Frag bfr;
            int col = wn * 64 + j * 16 + n;
            bfr.u[0] = *(const uint4*)&Xl[col][kbA];
            bfr.u[1] = *(const uint4*)&Xl[col][kbA + 16];
            acc[j] = bf16mma(a.v, bfr.v, acc[j]);
        }
    }

    // ---- epilogue: bias (+ optional scale) + layout-specific bf16 store ----
    const int dd0 = wm * 16 + Moff;                 // dd of row r=0 (multiple of 8)
    const size_t bh = (size_t)b * NH + blockIdx.x;  // head = blockIdx.x
    const float* bp = &bias[obase + dd0];
    float4 b0 = *(const float4*)bp;
    float4 b1 = *(const float4*)(bp + 4);

    if (MODE <= 1) {
        const float sc = (MODE == 0) ? SCALE : 1.0f;
        #pragma unroll
        for (int j = 0; j < 4; ++j) {
            int t = tbase + wn * 64 + j * 16 + n;
            union { uint4 u; v2bf h[4]; } pk;
            pk.h[0] = cvt2((acc[j][0] + b0.x) * sc, (acc[j][1] + b0.y) * sc);
            pk.h[1] = cvt2((acc[j][2] + b0.z) * sc, (acc[j][3] + b0.w) * sc);
            pk.h[2] = cvt2((acc[j][4] + b1.x) * sc, (acc[j][5] + b1.y) * sc);
            pk.h[3] = cvt2((acc[j][6] + b1.z) * sc, (acc[j][7] + b1.w) * sc);
            *(uint4*)&out[(bh * (size_t)T1 + t) * HD + dd0] = pk.u;   // 8 bf16, one b128
        }
    } else {
        float bb[8] = { b0.x, b0.y, b0.z, b0.w, b1.x, b1.y, b1.z, b1.w };
        #pragma unroll
        for (int j = 0; j < 4; ++j) {
            int t = tbase + wn * 64 + j * 16 + n;
            #pragma unroll
            for (int r = 0; r < 8; r += 2) {
                v2bf p = cvt2(acc[j][r] + bb[r], acc[j][r + 1] + bb[r + 1]);
                out[(bh * HD + dd0 + r    ) * (size_t)T2 + t] = p[0];
                out[(bh * HD + dd0 + r + 1) * (size_t)T2 + t] = p[1];
            }
        }
    }
}

// =====================================================================
// Flash attention: block = (b,h, 128 q rows); 8 waves x 16 q rows.
// k loop step 64 (4 score tiles); online softmax with WMMA row-sums.
// Double-buffered K/V tiles fed by global_load_async_to_lds_b128 (ASYNCcnt).
// K stored (t2,d): LDS tile Kl[k][d]  -> score B-frags contiguous
// V stored (d,t2): LDS tile Vl[d][k]  -> AV    B-frags contiguous
// =====================================================================
__global__ __launch_bounds__(256)
void attn_wmma_kernel(const __bf16* __restrict__ Qbf, const __bf16* __restrict__ Kbf,
                      const __bf16* __restrict__ Vbf,
                      const unsigned char* __restrict__ kvmask,
                      const float* __restrict__ x, float* __restrict__ out)
{
    const int h  = blockIdx.y;
    const int b  = blockIdx.z;
    const size_t bh = (size_t)b * NH + h;
    const int tid  = threadIdx.x;
    const int lane = tid & 31;
    const int wave = tid >> 5;
    const int qbase = blockIdx.x * 128 + wave * 16;

    __shared__ __align__(16) __bf16 Kl[2][64][64 + PADK];     // (k , d) x2
    __shared__ __align__(16) __bf16 Vl[2][64][64 + PADV];     // (d , k) x2
    __shared__ __align__(16) __bf16 Pl[8][16][64 + PADP];

    const int n    = lane & 15;
    const int rowb = (lane < 16) ? 0 : 16;   // B fragment K-dim base
    const int kbA  = (lane < 16) ? 0 : 8;    // A fragment K base
    const int Moff = (lane < 16) ? 0 : 8;    // C fragment M offset

    // Q fragments (two 16x32 A-frags covering d=0..63), loaded once
    v16bf qa0, qa1;
    {
        const __bf16* qp = &Qbf[(bh * T1 + qbase + n) * HD];
        Frag f0, f1;
        f0.u[0] = *(const uint4*)&qp[kbA];
        f0.u[1] = *(const uint4*)&qp[kbA + 16];
        f1.u[0] = *(const uint4*)&qp[32 + kbA];
        f1.u[1] = *(const uint4*)&qp[32 + kbA + 16];
        qa0 = f0.v; qa1 = f1.v;
    }

    // constant all-ones bf16 B fragment (1.0bf16 = 0x3F80) for row sums
    Frag ones;
    ones.u[0] = uint4{0x3F803F80u, 0x3F803F80u, 0x3F803F80u, 0x3F803F80u};
    ones.u[1] = ones.u[0];

    v8f oacc[4] = {};
    float m[8], l[8];
    #pragma unroll
    for (int r = 0; r < 8; ++r) { m[r] = -__builtin_inff(); l[r] = 0.f; }

    const int rT = tid >> 2, cgT = (tid & 3) * 16;   // tile coords (64 rows x 64 cols)

    // async DMA of one 64k x 64d K tile + 64d x 64k V tile into buffer p
    auto issue_tile = [&](int k0, int p) {
        const __bf16* ks = &Kbf[(bh * T2 + k0 + rT) * HD + cgT];
        const __bf16* vs = &Vbf[(bh * HD + rT) * T2 + k0 + cgT];
        async_b128(&Kl[p][rT][cgT],     ks);
        async_b128(&Kl[p][rT][cgT + 8], ks + 8);
        async_b128(&Vl[p][rT][cgT],     vs);
        async_b128(&Vl[p][rT][cgT + 8], vs + 8);
    };

    issue_tile(0, 0);   // prologue

    for (int k0 = 0; k0 < T2; k0 += 64) {
        const int p = (k0 >> 6) & 1;
        wait_async0();        // our slice of buffer p has landed
        __syncthreads();      // everyone's slice has landed; prev reads of p^1 done
        if (k0 + 64 < T2) issue_tile(k0 + 64, p ^ 1);   // overlap next DMA with compute

        // ---- scores: four 16x16 tiles over this 64-wide k strip ----
        v8f s[4] = {};
        #pragma unroll
        for (int dh = 0; dh < 2; ++dh) {
            v16bf aa = dh ? qa1 : qa0;
            int dbase = dh * 32 + kbA;
            #pragma unroll
            for (int j = 0; j < 4; ++j) {
                Frag kb;
                kb.u[0] = *(const uint4*)&Kl[p][j * 16 + n][dbase];
                kb.u[1] = *(const uint4*)&Kl[p][j * 16 + n][dbase + 16];
                s[j] = bf16mma(aa, kb.v, s[j]);
            }
        }

        // ---- kv_mask: masked column -> -inf ----
        #pragma unroll
        for (int j = 0; j < 4; ++j) {
            bool mk = kvmask[(size_t)b * T2 + k0 + j * 16 + n] != 0;
            if (!mk) {
                #pragma unroll
                for (int r = 0; r < 8; ++r) s[j][r] = -__builtin_inff();
            }
        }

        // ---- online softmax: row max via half-wave butterfly ----
        float alpha[8];
        #pragma unroll
        for (int r = 0; r < 8; ++r) {
            float mx = fmaxf(fmaxf(s[0][r], s[1][r]), fmaxf(s[2][r], s[3][r]));
            #pragma unroll
            for (int off = 1; off < 16; off <<= 1)
                mx = fmaxf(mx, __shfl_xor(mx, off, 32));
            float mn = fmaxf(m[r], mx);
            alpha[r] = __expf(m[r] - mn);
            m[r] = mn;
            #pragma unroll
            for (int j = 0; j < 4; ++j) s[j][r] = __expf(s[j][r] - mn);
        }
        #pragma unroll
        for (int dt = 0; dt < 4; ++dt)
            #pragma unroll
            for (int r = 0; r < 8; ++r) oacc[dt][r] *= alpha[r];

        // ---- re-layout P (C-layout f32 -> A-frag bf16) via per-wave LDS ----
        #pragma unroll
        for (int r = 0; r < 8; ++r) {
            v2bf p01 = cvt2(s[0][r], s[1][r]);
            v2bf p23 = cvt2(s[2][r], s[3][r]);
            Pl[wave][r + Moff][n]      = p01[0];
            Pl[wave][r + Moff][16 + n] = p01[1];
            Pl[wave][r + Moff][32 + n] = p23[0];
            Pl[wave][r + Moff][48 + n] = p23[1];
        }
        asm volatile("s_wait_dscnt 0" ::: "memory");
        Frag pa0, pa1;
        pa0.u[0] = *(const uint4*)&Pl[wave][n][kbA];
        pa0.u[1] = *(const uint4*)&Pl[wave][n][kbA + 16];
        pa1.u[0] = *(const uint4*)&Pl[wave][n][32 + kbA];
        pa1.u[1] = *(const uint4*)&Pl[wave][n][32 + kbA + 16];

        // ---- row sums of P via WMMA with all-ones B (replaces sum butterfly) ----
        v8f lsum = {};
        lsum = bf16mma(pa0.v, ones.v, lsum);
        lsum = bf16mma(pa1.v, ones.v, lsum);
        #pragma unroll
        for (int r = 0; r < 8; ++r) l[r] = l[r] * alpha[r] + lsum[r];

        // ---- O += P @ V ----
        #pragma unroll
        for (int dt = 0; dt < 4; ++dt) {
            Frag vb0, vb1;
            vb0.u[0] = *(const uint4*)&Vl[p][dt * 16 + n][kbA];
            vb0.u[1] = *(const uint4*)&Vl[p][dt * 16 + n][kbA + 16];
            vb1.u[0] = *(const uint4*)&Vl[p][dt * 16 + n][32 + kbA];
            vb1.u[1] = *(const uint4*)&Vl[p][dt * 16 + n][32 + kbA + 16];
            oacc[dt] = bf16mma(pa0.v, vb0.v, oacc[dt]);
            oacc[dt] = bf16mma(pa1.v, vb1.v, oacc[dt]);
        }
    }

    // ---- epilogue: O / l + residual x ; 8 consecutive q per lane -> float4 x2 ----
    float linv[8];
    #pragma unroll
    for (int r = 0; r < 8; ++r) linv[r] = 1.0f / l[r];

    #pragma unroll
    for (int dt = 0; dt < 4; ++dt) {
        int c = h * HD + dt * 16 + n;
        size_t base = ((size_t)b * QD + c) * (size_t)T1 + qbase + Moff;
        float4 x0 = *(const float4*)&x[base];
        float4 x1 = *(const float4*)&x[base + 4];
        float4 o0, o1;
        o0.x = oacc[dt][0] * linv[0] + x0.x;
        o0.y = oacc[dt][1] * linv[1] + x0.y;
        o0.z = oacc[dt][2] * linv[2] + x0.z;
        o0.w = oacc[dt][3] * linv[3] + x0.w;
        o1.x = oacc[dt][4] * linv[4] + x1.x;
        o1.y = oacc[dt][5] * linv[5] + x1.y;
        o1.z = oacc[dt][6] * linv[6] + x1.z;
        o1.w = oacc[dt][7] * linv[7] + x1.w;
        *(float4*)&out[base]     = o0;
        *(float4*)&out[base + 4] = o1;
    }
}

// mask passthrough (second tuple output)
__global__ void mask_copy_kernel(const unsigned char* __restrict__ m,
                                 float* __restrict__ out, int nmask)
{
    int i = blockIdx.x * 256 + threadIdx.x;
    if (i < nmask) out[i] = m[i] ? 1.0f : 0.0f;
}

extern "C" void kernel_launch(void* const* d_in, const int* in_sizes, int n_in,
                              void* d_out, int out_size, void* d_ws, size_t ws_size,
                              hipStream_t stream)
{
    const float*         x      = (const float*)d_in[0];
    const unsigned char* mask   = (const unsigned char*)d_in[1];
    const float*         kv     = (const float*)d_in[2];
    const unsigned char* kvmask = (const unsigned char*)d_in[3];
    const float* Wq = (const float*)d_in[4];
    const float* bq = (const float*)d_in[5];
    const float* Wk = (const float*)d_in[6];
    const float* bk = (const float*)d_in[7];
    const float* Wv = (const float*)d_in[8];
    const float* bv = (const float*)d_in[9];

    const size_t QKV_ELEMS = (size_t)BS * NH * T1 * HD;   // 8,388,608 each
    __bf16* Qbf = (__bf16*)d_ws;
    __bf16* Kbf = Qbf + QKV_ELEMS;
    __bf16* Vbf = Kbf + QKV_ELEMS;

    dim3 pb(256);
    dim3 pg(QD / 64, T1 / 128, BS);
    proj_wmma_kernel<0><<<pg, pb, 0, stream>>>(Wq, x,  bq, Qbf);
    proj_wmma_kernel<1><<<pg, pb, 0, stream>>>(Wk, kv, bk, Kbf);
    proj_wmma_kernel<2><<<pg, pb, 0, stream>>>(Wv, kv, bv, Vbf);

    dim3 ag(T1 / 128, NH, BS);
    attn_wmma_kernel<<<ag, pb, 0, stream>>>(Qbf, Kbf, Vbf, kvmask, x, (float*)d_out);

    const int out_elems = BS * QD * T1;                   // 8,388,608
    const int nmask = BS * 1 * T1;                        // 8192
    if (out_size >= out_elems + nmask)
        mask_copy_kernel<<<(nmask + 255) / 256, 256, 0, stream>>>(
            mask, (float*)d_out + out_elems, nmask);
}
